// AltRobertaSelfAttention_49228915146966
// MI455X (gfx1250) — compile-verified, matrix operand order checked by reading, not run on hardware
//
#include <hip/hip_runtime.h>

// ---------------------------------------------------------------------------
// CDNA5 (gfx1250) fused RoBERTa relative_key_query self-attention.
//  - bf16 WMMA (v_wmma_f32_16x16x32_bf16) for all GEMMs
//  - Tensor Data Mover (tensor_load_to_lds + s_wait_tensorcnt) for all tile
//    staging into LDS
//  - all operand tiles laid out so every WMMA fragment load is a pair of
//    contiguous ds_load_b128 (weights pre-transposed, V stored d-major)
//  - flash-attention tiling; positional bias as tile-local GEMMs against
//    dist_emb diagonal slices
// ---------------------------------------------------------------------------

typedef __attribute__((ext_vector_type(16))) __bf16 v16bf;
typedef __attribute__((ext_vector_type(8)))  float  v8f;
typedef unsigned int v4u __attribute__((ext_vector_type(4)));
typedef int          v8i __attribute__((ext_vector_type(8)));
typedef int          v4i __attribute__((ext_vector_type(4)));

constexpr int B_   = 4;
constexpr int S_   = 1024;
constexpr int HID_ = 1024;
constexpr int H_   = 16;
constexpr int HD_  = 64;

constexpr size_t NX = (size_t)B_ * S_ * HID_;     // 4194304  X elems
constexpr size_t NW = (size_t)HID_ * HID_;        // 1048576  per-proj weight elems
constexpr size_t NE = (size_t)2048 * 64;          // 131072   padded dist_emb elems
constexpr size_t NQ = (size_t)B_ * H_ * S_ * HD_; // 4194304  per Q/K/V

__device__ __forceinline__ v8f wmma_bf16(v16bf a, v16bf b, v8f c) {
  // (neg_a, A, neg_b, B, c_mod, C, reuse_a, reuse_b)
  return __builtin_amdgcn_wmma_f32_16x16x32_bf16(false, a, false, b, (short)0, c,
                                                 false, false);
}

// ---------------------------------------------------------------------------
// Tensor Data Mover: 2-D tile (bf16) global -> LDS, one DMA per call.
// D# layout per CDNA5 ISA 8.3/8.4: group0 = {flags, lds_addr, gaddr lo/hi+type},
// group1 = {mask/data_size, tensor dims, tile dims, dim0 stride}.
// Tensor dims are set equal to tile dims (tile fully in-bounds; stride carries
// the real row pitch).  Issued wave-uniform; tracked by TENSORcnt.
// ---------------------------------------------------------------------------
__device__ __forceinline__ void tdm_load_2d(unsigned lds_byte_off, const void* gptr,
                                            unsigned tile_d0, unsigned tile_d1,
                                            unsigned stride0_elems) {
  unsigned long long ga = (unsigned long long)(size_t)gptr;
  v4u g0;
  g0[0] = 1u;                                           // count=1, user descriptor
  g0[1] = lds_byte_off;                                 // LDS byte address
  g0[2] = (unsigned)ga;                                 // global addr [31:0]
  g0[3] = (unsigned)((ga >> 32) & 0x01FFFFFFull) | (2u << 30);  // [56:32] | type=2
  v8i g1;
  g1[0] = (int)(1u << 16);                              // wg_mask=0, data_size=1 (2B)
  g1[1] = (int)((tile_d0 & 0xFFFFu) << 16);             // tensor_dim0[15:0]
  g1[2] = (int)((tile_d1 & 0xFFFFu) << 16);             // tensor_dim0 hi=0 | tensor_dim1[15:0]
  g1[3] = (int)((tile_d0 & 0xFFFFu) << 16);             // tensor_dim1 hi=0 | tile_dim0
  g1[4] = (int)(tile_d1 & 0xFFFFu);                     // tile_dim1 | tile_dim2=0
  g1[5] = (int)stride0_elems;                           // tensor_dim0_stride[31:0]
  g1[6] = 0;                                            // stride0 hi | stride1 lo
  g1[7] = 0;                                            // stride1 hi
  v4i z4 = (v4i){0, 0, 0, 0};                           // groups 2/3 unused (2-D)
  v8i z8 = (v8i){0, 0, 0, 0, 0, 0, 0, 0};               // extra group (unused)
  __builtin_amdgcn_tensor_load_to_lds(g0, g1, z4, z4, z8, 0);
}

// Load a 16x16x32 A/B fragment from a row-major tile.
// rowptr points at this lane's row (A: m = lane&15, B: n = lane&15), 32 K-elems.
// h = lane>>4. Pairs p<4 hold K = 2p+8h; p>=4 hold K = 16+2(p-4)+8h
// => two contiguous 8-elem chunks at K = 8h and K = 16+8h  (2x ds_load_b128).
__device__ __forceinline__ v16bf load_frag(const __bf16* rowptr, int h) {
  v16bf f;
  const __bf16* p0 = rowptr + 8 * h;
  const __bf16* p1 = rowptr + 16 + 8 * h;
#pragma unroll
  for (int i = 0; i < 8; ++i) { f[i] = p0[i]; f[i + 8] = p1[i]; }
  return f;
}

// Store a 16x16 f32 C/D fragment to a row-major f32 LDS tile.
// C layout: lane L holds n = L&15; VGPR v holds row m = v + 8*(L>>4).
__device__ __forceinline__ void store_cfrag(float* dst, int ld, int row0, int col0,
                                            v8f c, int lane) {
  int n = lane & 15, hh = lane >> 4;
#pragma unroll
  for (int v = 0; v < 8; ++v)
    dst[(row0 + v + 8 * hh) * ld + col0 + n] = c[v];
}

__device__ __forceinline__ float hmax16(float x) {
#pragma unroll
  for (int m = 1; m < 16; m <<= 1) x = fmaxf(x, __shfl_xor(x, m, 32));
  return x;
}
__device__ __forceinline__ float hsum16(float x) {
#pragma unroll
  for (int m = 1; m < 16; m <<= 1) x += __shfl_xor(x, m, 32);
  return x;
}

// ---------------------------------------------------------------------------
// Kernel 1: f32 -> bf16 conversion.
//   X  -> Xb   [4096][1024]          (row-major, unchanged)
//   Wq|Wk|Wv -> Wb TRANSPOSED: Wb[p][n][k] = W[k][n]   (B-operand friendly)
//   dist_emb -> Eb, padded to 2048 rows
// ---------------------------------------------------------------------------
__global__ __launch_bounds__(256) void convert_kernel(
    const float* __restrict__ X, const float* __restrict__ Wq,
    const float* __restrict__ Wk, const float* __restrict__ Wv,
    const float* __restrict__ E, __bf16* __restrict__ Xb,
    __bf16* __restrict__ Wb, __bf16* __restrict__ Eb) {
  size_t i = (size_t)blockIdx.x * 256 + threadIdx.x;
  if (i < NX) {
    Xb[i] = (__bf16)X[i];
    return;
  }
  i -= NX;
  if (i < 3 * NW) {
    size_t p = i / NW, j = i % NW;
    const float* W = (p == 0) ? Wq : (p == 1) ? Wk : Wv;
    size_t k = j >> 10, n = j & 1023;           // read coalesced over n
    Wb[p * NW + n * HID_ + k] = (__bf16)W[j];   // store transposed
    return;
  }
  i -= 3 * NW;
  if (i < NE) {
    Eb[i] = (i < (size_t)2047 * 64) ? (__bf16)E[i] : (__bf16)0.0f;
  }
}

// ---------------------------------------------------------------------------
// Kernel 2: fused QKV projection GEMM.
// Y[4096,3072] = Xb[4096,1024] @ [Wq|Wk|Wv] + bias.
// Q,K scattered to [B,H,S,hd]; V scattered to [B,H,hd,S] (d-major, so the
// attention kernel's P@V B-fragments are contiguous).
// Block = 128 thr (4 waves); block tile 64 rows x 64 cols; wave tile 16x64.
// A/B tiles staged by TDM; B tile already transposed ([n][k]).
// ---------------------------------------------------------------------------
__global__ __launch_bounds__(128) void qkv_kernel(
    const __bf16* __restrict__ Xb, const __bf16* __restrict__ Wb,
    const float* __restrict__ bq, const float* __restrict__ bk,
    const float* __restrict__ bv, __bf16* __restrict__ Qb,
    __bf16* __restrict__ Kb, __bf16* __restrict__ Vb) {
  extern __shared__ char smem[];
  __bf16* sA = (__bf16*)smem;              // [m=64][k=32]
  __bf16* sB = sA + 64 * 32;               // [n=64][k=32] (row-major, TDM order)
  const unsigned offA = 0;
  const unsigned offB = 64 * 32 * sizeof(__bf16);

  const int tid = threadIdx.x;
  const int wave = tid >> 5, lane = tid & 31;
  const int n = lane & 15, h = lane >> 4, hh = h;

  const int row0 = blockIdx.x * 64;          // over 4096 rows
  const int c0 = blockIdx.y * 64;            // over 3072 cols
  const int proj = c0 >> 10;                 // 0:Q 1:K 2:V (uniform per block)
  const int cc0 = c0 & 1023;
  const __bf16* Wt = Wb + (size_t)proj * NW; // [n][k] transposed weights

  v8f acc[4];
#pragma unroll
  for (int t = 0; t < 4; ++t) acc[t] = (v8f){};

  for (int k0 = 0; k0 < HID_; k0 += 32) {
    __syncthreads();
    if (wave == 0) {
      // A: 64 m-rows x 32 k-elems, row pitch 1024
      tdm_load_2d(offA, Xb + (size_t)row0 * HID_ + k0, 32, 64, HID_);
      // B: 64 n-rows x 32 k-elems, row pitch 1024 (transposed weights)
      tdm_load_2d(offB, Wt + (size_t)cc0 * HID_ + k0, 32, 64, HID_);
      __builtin_amdgcn_s_wait_tensorcnt(0);
    }
    __syncthreads();

    v16bf a = load_frag(sA + (wave * 16 + n) * 32, h);
#pragma unroll
    for (int nt = 0; nt < 4; ++nt) {
      v16bf b = load_frag(sB + (nt * 16 + n) * 32, h);
      acc[nt] = wmma_bf16(a, b, acc[nt]);
    }
  }

  // Epilogue: bias add, bf16 convert, scatter.
  const float* bias = (proj == 0) ? bq : (proj == 1) ? bk : bv;
  __bf16* dst = (proj == 0) ? Qb : (proj == 1) ? Kb : Vb;
#pragma unroll
  for (int nt = 0; nt < 4; ++nt) {
    int cc = cc0 + nt * 16 + n;
    float bval = bias[cc];
    int head = cc >> 6, d = cc & 63;
#pragma unroll
    for (int v = 0; v < 8; ++v) {
      int row = row0 + wave * 16 + v + 8 * hh;   // global token row
      int bb = row >> 10, ss = row & 1023;
      __bf16 val = (__bf16)(acc[nt][v] + bval);
      if (proj != 2)  // Q,K: [bh][s][d]
        dst[(((size_t)(bb * H_ + head)) * S_ + ss) * HD_ + d] = val;
      else            // V:   [bh][d][s]  (d-major)
        dst[(((size_t)(bb * H_ + head)) * HD_ + d) * S_ + ss] = val;
    }
  }
}

// ---------------------------------------------------------------------------
// Kernel 3: flash attention with relative_key_query bias.
//   grid = (S/64, B*H); block = 128 thr (4 waves), 64 query rows per block,
//   wave w owns query rows [16w, 16w+16).
// Per 64-wide key tile (K/V/E staged by TDM):
//   S  = Qt Kt^T                                (WMMA)
//   QE = Qt Eslice^T, KE = Kt Eslice^T          (WMMA, Eslice = 128 dist_emb rows)
//   bias[l,r] = QE[l, l-r+63] + KE[r, l-r+63]   (LDS gather)
//   online softmax, acc += P V                  (WMMA, V tile is [d][r])
// ---------------------------------------------------------------------------
__global__ __launch_bounds__(128) void attn_kernel(
    const __bf16* __restrict__ Qb, const __bf16* __restrict__ Kb,
    const __bf16* __restrict__ Vt, const __bf16* __restrict__ Eb,
    const float* __restrict__ mask, float* __restrict__ out) {
  extern __shared__ char smem[];
  __bf16* sQ = (__bf16*)smem;               // 64 x 64   [l][d]
  __bf16* sK = sQ + 64 * 64;                // 64 x 64   [r][d]
  __bf16* sV = sK + 64 * 64;                // 64 x 64   [d][r]  (d-major)
  __bf16* sE = sV + 64 * 64;                // 128 x 64  [j][d]
  __bf16* sP = sE + 128 * 64;               // 4 waves x 16 x 64
  float* sQE = (float*)(sP + 4 * 16 * 64);  // 64 x 128
  float* sKE = sQE + 64 * 128;              // 64 x 128
  const unsigned offQ = 0;
  const unsigned offK = 64 * 64 * 2;
  const unsigned offV = 2 * 64 * 64 * 2;
  const unsigned offE = 3 * 64 * 64 * 2;

  const int tid = threadIdx.x;
  const int wave = tid >> 5, lane = tid & 31;
  const int n = lane & 15, h = lane >> 4, hh = h;

  const int bh = blockIdx.y;
  const int b = bh >> 4, head = bh & 15;
  const int l0 = blockIdx.x * 64;

  // Q tile once: 64 rows x 64 d, contiguous rows.
  if (wave == 0) {
    tdm_load_2d(offQ, Qb + ((size_t)bh * S_ + l0) * HD_, 64, 64, 64);
    __builtin_amdgcn_s_wait_tensorcnt(0);
  }
  __syncthreads();

  // Per-wave Q A-fragments (rows 16w .. 16w+15), K-chunks 0..31 and 32..63.
  v16bf aQ0 = load_frag(sQ + (wave * 16 + n) * HD_, h);
  v16bf aQ1 = load_frag(sQ + (wave * 16 + n) * HD_ + 32, h);

  float mrow[8], lrow[8];
  v8f acc[4];
#pragma unroll
  for (int v = 0; v < 8; ++v) { mrow[v] = -3.0e38f; lrow[v] = 0.0f; }
#pragma unroll
  for (int t = 0; t < 4; ++t) acc[t] = (v8f){};

  for (int r0 = 0; r0 < S_; r0 += 64) {
    __syncthreads();   // previous iteration fully consumed sK/sV/sE/sQE/sKE

    // ---- stage K, V, E-slice into LDS via TDM ----
    const int dmin = l0 - r0 + 960;   // delta = l-r+1023, tile min; in [0,1920]
    if (wave == 0) {
      tdm_load_2d(offK, Kb + ((size_t)bh * S_ + r0) * HD_, 64, 64, 64);
      // V is stored [bh][d][s]: 64 d-rows x 64 r-elems, row pitch S
      tdm_load_2d(offV, Vt + (size_t)bh * HD_ * S_ + r0, 64, 64, S_);
      tdm_load_2d(offE, Eb + (size_t)dmin * 64, 64, 128, 64);
      __builtin_amdgcn_s_wait_tensorcnt(0);
    }
    if (r0 + 64 < S_) {  // prefetch next K tile -> global_prefetch_b8
      const char* nk = (const char*)(Kb + ((size_t)bh * S_ + r0 + 64) * HD_);
      __builtin_prefetch(nk + tid * 64, 0, 1);
    }
    __syncthreads();

    // ---- S = Q K^T (16x64 per wave) ----
    v8f sS[4];
#pragma unroll
    for (int nt = 0; nt < 4; ++nt) {
      v8f c = (v8f){};
      v16bf bk0 = load_frag(sK + (nt * 16 + n) * HD_, h);
      v16bf bk1 = load_frag(sK + (nt * 16 + n) * HD_ + 32, h);
      c = wmma_bf16(aQ0, bk0, c);
      c = wmma_bf16(aQ1, bk1, c);
      sS[nt] = c;
    }

    // ---- QE (own rows) and KE (rows 16w..16w+15) vs 128 E-diagonals ----
    v16bf aK0 = load_frag(sK + (wave * 16 + n) * HD_, h);
    v16bf aK1 = load_frag(sK + (wave * 16 + n) * HD_ + 32, h);
#pragma unroll
    for (int jt = 0; jt < 8; ++jt) {
      v16bf be0 = load_frag(sE + (jt * 16 + n) * HD_, h);
      v16bf be1 = load_frag(sE + (jt * 16 + n) * HD_ + 32, h);
      v8f cq = (v8f){};
      cq = wmma_bf16(aQ0, be0, cq);
      cq = wmma_bf16(aQ1, be1, cq);
      v8f ck = (v8f){};
      ck = wmma_bf16(aK0, be0, ck);
      ck = wmma_bf16(aK1, be1, ck);
      store_cfrag(sQE, 128, wave * 16, jt * 16, cq, lane);
      store_cfrag(sKE, 128, wave * 16, jt * 16, ck, lane);
    }
    __syncthreads();   // KE from all waves visible

    // ---- bias gather + scale + mask, tile row-max ----
    float tmax[8];
#pragma unroll
    for (int v = 0; v < 8; ++v) tmax[v] = -3.0e38f;
#pragma unroll
    for (int nt = 0; nt < 4; ++nt) {
      int rn = nt * 16 + n;
      float mval = mask[(size_t)b * S_ + r0 + rn];
#pragma unroll
      for (int v = 0; v < 8; ++v) {
        int lm = wave * 16 + v + 8 * hh;       // local query row in [0,64)
        int j = lm - rn + 63;                  // diagonal index in [0,127)
        float s = sS[nt][v] + sQE[lm * 128 + j] + sKE[rn * 128 + j];
        s = s * 0.125f + mval;                 // 1/sqrt(64)
        sS[nt][v] = s;
        tmax[v] = fmaxf(tmax[v], s);
      }
    }
#pragma unroll
    for (int v = 0; v < 8; ++v) tmax[v] = hmax16(tmax[v]);

    // ---- online softmax update ----
#pragma unroll
    for (int v = 0; v < 8; ++v) {
      float mnew = fmaxf(mrow[v], tmax[v]);
      float corr = __expf(mrow[v] - mnew);
      mrow[v] = mnew;
      lrow[v] *= corr;
#pragma unroll
      for (int nt = 0; nt < 4; ++nt) acc[nt][v] *= corr;
    }
#pragma unroll
    for (int nt = 0; nt < 4; ++nt)
#pragma unroll
      for (int v = 0; v < 8; ++v) sS[nt][v] = __expf(sS[nt][v] - mrow[v]);
#pragma unroll
    for (int v = 0; v < 8; ++v) {
      float rs = 0.0f;
#pragma unroll
      for (int nt = 0; nt < 4; ++nt) rs += sS[nt][v];
      lrow[v] += hsum16(rs);
    }

    // ---- P -> LDS (wave-private, DS ops in-order within wave), P @ V ----
    __bf16* sPw = sP + wave * 16 * 64;
#pragma unroll
    for (int nt = 0; nt < 4; ++nt)
#pragma unroll
      for (int v = 0; v < 8; ++v)
        sPw[(v + 8 * hh) * 64 + nt * 16 + n] = (__bf16)sS[nt][v];

    v16bf aP0 = load_frag(sPw + n * 64, h);
    v16bf aP1 = load_frag(sPw + n * 64 + 32, h);
#pragma unroll
    for (int dt = 0; dt < 4; ++dt) {
      // sV is [d][r]: row = d = dt*16+n, K = r (chunks 0..31, 32..63)
      v16bf bv0 = load_frag(sV + (dt * 16 + n) * 64, h);
      v16bf bv1 = load_frag(sV + (dt * 16 + n) * 64 + 32, h);
      acc[dt] = wmma_bf16(aP0, bv0, acc[dt]);
      acc[dt] = wmma_bf16(aP1, bv1, acc[dt]);
    }
  }

  // ---- epilogue: normalize and write ctx as [B, S, H*hd] f32 ----
#pragma unroll
  for (int v = 0; v < 8; ++v) {
    float inv = 1.0f / lrow[v];
    int l = l0 + wave * 16 + v + 8 * hh;
#pragma unroll
    for (int dt = 0; dt < 4; ++dt) {
      int d = dt * 16 + n;
      out[((size_t)(b * S_ + l)) * HID_ + head * HD_ + d] = acc[dt][v] * inv;
    }
  }
}

// ---------------------------------------------------------------------------
// Host launcher
// ---------------------------------------------------------------------------
extern "C" void kernel_launch(void* const* d_in, const int* in_sizes, int n_in,
                              void* d_out, int out_size, void* d_ws, size_t ws_size,
                              hipStream_t stream) {
  const float* X    = (const float*)d_in[0];
  const float* msk  = (const float*)d_in[1];
  const float* Wq   = (const float*)d_in[2];
  const float* bq   = (const float*)d_in[3];
  const float* Wk   = (const float*)d_in[4];
  const float* bk   = (const float*)d_in[5];
  const float* Wv   = (const float*)d_in[6];
  const float* bv   = (const float*)d_in[7];
  const float* dist = (const float*)d_in[8];
  float* out = (float*)d_out;

  // Workspace carve-out (all bf16): X 8MB, W 6MB, E 0.25MB, Q/K/V 8MB each.
  char* ws = (char*)d_ws;
  __bf16* Xb = (__bf16*)ws;            ws += NX * 2;
  __bf16* Wb = (__bf16*)ws;            ws += 3 * NW * 2;
  __bf16* Eb = (__bf16*)ws;            ws += NE * 2;
  __bf16* Qb = (__bf16*)ws;            ws += NQ * 2;
  __bf16* Kb = (__bf16*)ws;            ws += NQ * 2;
  __bf16* Vb = (__bf16*)ws;            ws += NQ * 2;

  // 1) convert f32 -> bf16 (X, transposed weights, padded dist_emb)
  {
    size_t total = NX + 3 * NW + NE;                 // 7,471,104
    int blocks = (int)((total + 255) / 256);
    convert_kernel<<<blocks, 256, 0, stream>>>(X, Wq, Wk, Wv, dist, Xb, Wb, Eb);
  }

  // 2) fused QKV projection (dynamic LDS: A 4KB + B 4KB)
  {
    dim3 grid(4096 / 64, 3072 / 64);                 // (64, 48)
    size_t smem = (size_t)(64 * 32 + 64 * 32) * sizeof(__bf16);
    qkv_kernel<<<grid, 128, smem, stream>>>(Xb, Wb, bq, bk, bv, Qb, Kb, Vb);
  }

  // 3) flash attention with relative-position bias
  {
    dim3 grid(S_ / 64, B_ * H_);                     // (16, 64)
    size_t smem = (size_t)(64 * 64 * 3 + 128 * 64 + 4 * 16 * 64) * sizeof(__bf16)
                + (size_t)(2 * 64 * 128) * sizeof(float);   // 114,688 B
    attn_kernel<<<grid, 128, smem, stream>>>(Qb, Kb, Vb, Eb, msk, out);
  }
}
